// GCN_23948737643064
// MI455X (gfx1250) — matrix operand order repaired and checked by reading, not compiled, not run on hardware
//
#include <hip/hip_runtime.h>
#include <hip/hip_bf16.h>
#include <math.h>

typedef __attribute__((ext_vector_type(16))) _Float16 v16h;
typedef __attribute__((ext_vector_type(8)))  float    v8f;

#define NNODES 100000   // multiple of 16 (6250 row tiles) -- no row tails
#define NEDGES 1600000

// ---------------------------------------------------------------- utilities
__global__ void k_fill(float* __restrict__ p, float v, int n) {
    int i = blockIdx.x * 256 + threadIdx.x;
    if (i < n) p[i] = v;
}

// deg[col] += 1 for each edge (deg pre-filled with 1.0 for self loop)
__global__ void k_deg(const int* __restrict__ col, float* __restrict__ deg, int E) {
    int i = blockIdx.x * 256 + threadIdx.x;
    if (i < E) atomicAdd(&deg[col[i]], 1.0f);
}

__global__ void k_rsqrt(float* __restrict__ d, int n) {
    int i = blockIdx.x * 256 + threadIdx.x;
    if (i < n) {
        float v = d[i];
        d[i] = v > 0.0f ? rsqrtf(v) : 0.0f;
    }
}

// -------------------------------------------------- weight fragment packing
// Pack W[128 x OUT] (f32) into per-lane WMMA B fragments (f16):
//   Wp[((t*4 + s)*32 + lane)*16 + e]  with  t = col tile, s = k-step (32 K),
//   n = t*16 + lane%16, g = lane/16, k = s*32 + 16*g + 2*(e/2) + (e&1).
// Each lane's 16 halves are 32 contiguous bytes -> two global_load_b128.
template<int OUT>
__global__ void k_packW(const float* __restrict__ W, _Float16* __restrict__ Wp)
{
    constexpr int NT = (OUT + 15) / 16;
    int i = blockIdx.x * 256 + threadIdx.x;
    if (i >= NT * 2048) return;               // 2048 = 4 ksteps * 32 lanes * 16
    int e    = i & 15;
    int lane = (i >> 4) & 31;
    int s    = (i >> 9) & 3;
    int t    = i >> 11;
    int n = t * 16 + (lane & 15);
    int g = lane >> 4;
    int k = s * 32 + 16 * g + 2 * (e >> 1) + (e & 1);
    float v = (n < OUT) ? W[k * OUT + n] : 0.0f;
    Wp[i] = (_Float16)v;
}

// ------------------------------------------------------------ WMMA GEMM
// H[nrows x OUT] = act(A[nrows x 128]) @ W[128 x OUT]
// act = identity (BN=false) or relu(A*scale+shift) (BN=true, fused BatchNorm)
// One workgroup = 16 rows; 8 waves, wave w owns 16-col tile w.
// PRECONDITION (enforced by launcher): nrows is a multiple of 16 and the grid
// is exactly nrows/16 blocks -> no row bounds checks anywhere.
template<int OUT, bool BN>
__global__ __launch_bounds__(256)
void k_gemm(const float* __restrict__ A, const _Float16* __restrict__ Wp,
            const float* __restrict__ scale, const float* __restrict__ shift,
            float* __restrict__ Hout)
{
    constexpr int K = 128;
    constexpr int LDA = 136;                 // padded LDS row stride (halfs)
    __shared__ _Float16 As[16 * LDA];

    const int tid     = threadIdx.x;
    const int rowBase = blockIdx.x * 16;

    // Cooperative stage of A tile (16 x 128) to LDS as f16, BN+ReLU fused.
    for (int i = tid; i < 16 * K; i += 256) {
        int r = i >> 7;
        int k = i & 127;
        float v = A[(rowBase + r) * K + k];
        if constexpr (BN) {
            v = v * scale[k] + shift[k];
            v = v > 0.0f ? v : 0.0f;         // ReLU
        }
        As[r * LDA + k] = (_Float16)v;
    }
    __syncthreads();

    const int wave = tid >> 5;
    const int lane = tid & 31;
    constexpr int NT = (OUT + 15) / 16;      // column tiles
    if (wave >= NT) return;

    const int colBase = wave * 16;
    const int m = lane & 15;                 // A row within tile / B,D col
    const int g = lane >> 4;                 // lane group
    const int n = colBase + m;               // global output column

    const _Float16* wbase = Wp + wave * 2048 + lane * 16;

    v8f c = {};
    #pragma unroll
    for (int s = 0; s < 4; ++s) {            // k-steps of 32
        const int kb = s * 32;
        // ---- A fragment: contiguous k order k = 16*g + e (matches packed B)
        union { v16h v; unsigned int u[8]; } a;
        #pragma unroll
        for (int vv = 0; vv < 8; ++vv) {
            int k0 = 2 * vv + g * 16; // even
            a.u[vv] = *(const unsigned int*)&As[m * LDA + kb + k0];
        }
        // ---- B fragment: pre-packed, 32 contiguous bytes per lane
        v16h b = *(const v16h*)(wbase + s * 512);
        c = __builtin_amdgcn_wmma_f32_16x16x32_f16(
                /*neg_a=*/false, a.v, /*neg_b=*/false, b,
                /*c_mod=*/(short)0, c, /*reuse_a=*/false, /*reuse_b=*/false);
    }

    // D store: VGPR r -> row r + 8*g, lane%16 -> col. 32-bit indexing.
    const int baseIdx = (rowBase + 8 * g) * OUT + n;
    if constexpr (OUT % 16 == 0) {
        #pragma unroll
        for (int r = 0; r < 8; ++r)
            Hout[baseIdx + r * OUT] = c[r];
    } else {
        if (n < OUT) {
            #pragma unroll
            for (int r = 0; r < 8; ++r)
                Hout[baseIdx + r * OUT] = c[r];
        }
    }
}

// ------------------------------------------------------------ edge scatter
// agg[col] += h[row] * dinv[row] * dinv[col]   (one wave per edge)
template<int OUT>
__global__ __launch_bounds__(256)
void k_scatter(const int* __restrict__ ei, const float* __restrict__ h,
               const float* __restrict__ dinv, float* __restrict__ agg, int E)
{
    int wid  = (blockIdx.x * 256 + threadIdx.x) >> 5;
    int lane = threadIdx.x & 31;
    int nw   = (gridDim.x * 256) >> 5;
    for (int e = wid; e < E; e += nw) {
        int row = ei[e];
        int col = ei[E + e];
        float nrm = dinv[row] * dinv[col];
        if (OUT == 128) {
            const float4 v = *(const float4*)&h[row * 128 + lane * 4];
            float* dst = &agg[col * 128 + lane * 4];
            atomicAdd(dst + 0, v.x * nrm);
            atomicAdd(dst + 1, v.y * nrm);
            atomicAdd(dst + 2, v.z * nrm);
            atomicAdd(dst + 3, v.w * nrm);
        } else {
            for (int f = lane; f < OUT; f += 32)
                atomicAdd(&agg[col * OUT + f], h[row * OUT + f] * nrm);
        }
    }
}

// self-loop contribution + bias:  agg[i,f] += h[i,f]*dinv[i]^2 + b[f]
template<int OUT>
__global__ void k_finalize(float* __restrict__ agg, const float* __restrict__ h,
                           const float* __restrict__ dinv, const float* __restrict__ b,
                           int nrows)
{
    int i = blockIdx.x * 256 + threadIdx.x;
    if (i < nrows * OUT) {
        int node = i / OUT;
        int f    = i - node * OUT;
        float d  = dinv[node];
        agg[i] += h[i] * d * d + b[f];
    }
}

// ------------------------------------------------------------ BatchNorm
// stats[0..127] = sum, stats[128..255] = sum of squares (atomic partials)
__global__ __launch_bounds__(256)
void k_bnstats(const float* __restrict__ h, float* __restrict__ stats, int nrows)
{
    int f    = threadIdx.x & 127;
    int half = threadIdx.x >> 7;
    float s = 0.0f, ss = 0.0f;
    for (int r = blockIdx.x * 2 + half; r < nrows; r += gridDim.x * 2) {
        float v = h[r * 128 + f];
        s  += v;
        ss += v * v;
    }
    __shared__ float ls[256], lss[256];
    ls[threadIdx.x]  = s;
    lss[threadIdx.x] = ss;
    __syncthreads();
    if (threadIdx.x < 128) {
        atomicAdd(&stats[f],       ls[threadIdx.x]  + ls[threadIdx.x + 128]);
        atomicAdd(&stats[128 + f], lss[threadIdx.x] + lss[threadIdx.x + 128]);
    }
}

__global__ void k_bnfinal(const float* __restrict__ stats,
                          const float* __restrict__ gamma, const float* __restrict__ beta,
                          float* __restrict__ scale, float* __restrict__ shift, int nrows)
{
    int f = threadIdx.x;
    if (f < 128) {
        float invN = 1.0f / (float)nrows;
        float mean = stats[f] * invN;
        float var  = stats[128 + f] * invN - mean * mean;
        float is   = rsqrtf(var + 1e-5f);
        float sc   = gamma[f] * is;
        scale[f] = sc;
        shift[f] = beta[f] - mean * sc;
    }
}

// --------------------------------------------- layer3 finalize + log_softmax
// out[i,:] = logsoftmax(agg[i,:] + h3[i,:]*dinv[i]^2 + b3)  over C=40
__global__ __launch_bounds__(256)
void k_lsm(const float* __restrict__ agg, const float* __restrict__ h3,
           const float* __restrict__ dinv, const float* __restrict__ b,
           float* __restrict__ out, int nrows)
{
    int wid  = (blockIdx.x * 256 + threadIdx.x) >> 5;
    int lane = threadIdx.x & 31;
    int nw   = (gridDim.x * 256) >> 5;
    for (int node = wid; node < nrows; node += nw) {
        float d  = dinv[node];
        float d2 = d * d;
        int f0 = lane, f1 = lane + 32;
        float v0 = -INFINITY, v1 = -INFINITY;
        if (f0 < 40) v0 = agg[node * 40 + f0] + h3[node * 40 + f0] * d2 + b[f0];
        if (f1 < 40) v1 = agg[node * 40 + f1] + h3[node * 40 + f1] * d2 + b[f1];
        float m = fmaxf(v0, v1);
        #pragma unroll
        for (int o = 16; o; o >>= 1) m = fmaxf(m, __shfl_xor(m, o, 32));
        float s = (f0 < 40 ? __expf(v0 - m) : 0.0f) + (f1 < 40 ? __expf(v1 - m) : 0.0f);
        #pragma unroll
        for (int o = 16; o; o >>= 1) s += __shfl_xor(s, o, 32);
        float lse = m + logf(s);
        if (f0 < 40) out[node * 40 + f0] = v0 - lse;
        if (f1 < 40) out[node * 40 + f1] = v1 - lse;
    }
}

// ================================================================ launcher
extern "C" void kernel_launch(void* const* d_in, const int* in_sizes, int n_in,
                              void* d_out, int out_size, void* d_ws, size_t ws_size,
                              hipStream_t stream)
{
    const float* x   = (const float*)d_in[0];
    const int*   ei  = (const int*)  d_in[1];
    const float* W1  = (const float*)d_in[2];
    const float* b1  = (const float*)d_in[3];
    const float* g1  = (const float*)d_in[4];
    const float* be1 = (const float*)d_in[5];
    const float* W2  = (const float*)d_in[6];
    const float* b2  = (const float*)d_in[7];
    const float* g2  = (const float*)d_in[8];
    const float* be2 = (const float*)d_in[9];
    const float* W3  = (const float*)d_in[10];
    const float* b3  = (const float*)d_in[11];
    float* out = (float*)d_out;

    const int N = NNODES, E = NEDGES;

    // ---- workspace carve-up (all 256B aligned)
    char* ws = (char*)d_ws;
    size_t off = 0;
    auto carve = [&](size_t bytes) -> void* {
        void* p = (void*)(ws + off);
        off = (off + bytes + 255) & ~(size_t)255;
        return p;
    };
    float*    dinv   = (float*)   carve((size_t)N * 4);
    float*    hbuf   = (float*)   carve((size_t)N * 128 * 4);   // h1 / h2 / h3
    float*    aggbuf = (float*)   carve((size_t)N * 128 * 4);   // agg1 / agg2 / agg3
    float*    stats  = (float*)   carve(256 * 4);
    float*    scale  = (float*)   carve(128 * 4);
    float*    shift  = (float*)   carve(128 * 4);
    _Float16* Wp     = (_Float16*)carve(8 * 2048 * 2);          // packed weights

    const int MT = N / 16;                        // 6250 row tiles (N % 16 == 0)

    // ---- normalization: deg (with self loop) -> dinv
    k_fill <<<(N + 255) / 256, 256, 0, stream>>>(dinv, 1.0f, N);
    k_deg  <<<(E + 255) / 256, 256, 0, stream>>>(ei + E, dinv, E);
    k_rsqrt<<<(N + 255) / 256, 256, 0, stream>>>(dinv, N);

    // ---- layer 1: h1 = x @ W1 ; agg1 = scatter(h1) + self + b1
    k_packW<128><<<(8 * 2048 + 255) / 256, 256, 0, stream>>>(W1, Wp);
    k_gemm<128, false><<<MT, 256, 0, stream>>>(x, Wp, nullptr, nullptr, hbuf);
    k_fill<<<(N * 128 + 255) / 256, 256, 0, stream>>>(aggbuf, 0.0f, N * 128);
    k_scatter<128><<<4096, 256, 0, stream>>>(ei, hbuf, dinv, aggbuf, E);
    k_finalize<128><<<(N * 128 + 255) / 256, 256, 0, stream>>>(aggbuf, hbuf, dinv, b1, N);
    k_fill<<<1, 256, 0, stream>>>(stats, 0.0f, 256);
    k_bnstats<<<1024, 256, 0, stream>>>(aggbuf, stats, N);
    k_bnfinal<<<1, 128, 0, stream>>>(stats, g1, be1, scale, shift, N);

    // ---- layer 2: h2 = relu(bn(agg1)) @ W2 (BN fused into A-stage)
    k_packW<128><<<(8 * 2048 + 255) / 256, 256, 0, stream>>>(W2, Wp);
    k_gemm<128, true><<<MT, 256, 0, stream>>>(aggbuf, Wp, scale, shift, hbuf);
    k_fill<<<(N * 128 + 255) / 256, 256, 0, stream>>>(aggbuf, 0.0f, N * 128);
    k_scatter<128><<<4096, 256, 0, stream>>>(ei, hbuf, dinv, aggbuf, E);
    k_finalize<128><<<(N * 128 + 255) / 256, 256, 0, stream>>>(aggbuf, hbuf, dinv, b2, N);
    k_fill<<<1, 256, 0, stream>>>(stats, 0.0f, 256);
    k_bnstats<<<1024, 256, 0, stream>>>(aggbuf, stats, N);
    k_bnfinal<<<1, 128, 0, stream>>>(stats, g2, be2, scale, shift, N);

    // ---- layer 3: h3 = relu(bn(agg2)) @ W3 ; scatter ; log_softmax
    k_packW<40><<<(3 * 2048 + 255) / 256, 256, 0, stream>>>(W3, Wp);
    k_gemm<40, true><<<MT, 256, 0, stream>>>(aggbuf, Wp, scale, shift, hbuf);
    k_fill<<<(N * 40 + 255) / 256, 256, 0, stream>>>(aggbuf, 0.0f, N * 40);
    k_scatter<40><<<4096, 256, 0, stream>>>(ei, hbuf, dinv, aggbuf, E);
    k_lsm<<<(N + 7) / 8, 256, 0, stream>>>(aggbuf, hbuf, dinv, b3, out, N);

    (void)in_sizes; (void)n_in; (void)out_size; (void)ws_size;
}
